// GraphAttentionLayer_81827716924076
// MI455X (gfx1250) — matrix operand order repaired and checked by reading, not compiled, not run on hardware
//
#include <hip/hip_runtime.h>

// ---------------------------------------------------------------------------
// GraphAttentionLayer for MI455X (gfx1250, wave32, WMMA + async global->LDS)
//   h = x @ W^T ; p = h@a1 ; q = h@a2
//   s_ij = LeakyReLU(p_i + q_j); attn = softmax_j(s); out = attn @ h
// LeakyReLU monotone => rowmax_i = LR(p_i + max_j q_j): analytic softmax max,
// so the big attn@h pass is a clean streaming bf16-WMMA GEMM with exp-weights
// built in registers. hb j-tiles are DMA'd to LDS with
// global_load_async_to_lds_b128 (ASYNCcnt) and double-buffered.
// ---------------------------------------------------------------------------

typedef __bf16 bf16;
typedef __attribute__((ext_vector_type(16))) __bf16 v16bf;
typedef __attribute__((ext_vector_type(8)))  __bf16 v8bf;
typedef __attribute__((ext_vector_type(8)))  float  v8f;
typedef __attribute__((ext_vector_type(4)))  float  v4f;

#define NROWS 8192
#define FIN   512
#define FOUT  512
#define ALPHA 0.2f

#define JSPLIT   2
#define JTILE    32
#define ROWS_BLK 64
#define LDS_PAD  8
#define LDS_STRIDE (FOUT + LDS_PAD)   // 520 bf16 per row (bank-shifted)

__device__ __forceinline__ float lrelu(float s) { return s >= 0.f ? s : ALPHA * s; }

// async DMA: one instruction moves 16B per lane (512B per wave) into LDS.
__device__ __forceinline__ void async_ld_b128(unsigned lds_off, const bf16* g) {
  asm volatile("global_load_async_to_lds_b128 %0, %1, off"
               :: "v"(lds_off), "v"(g) : "memory");
}
__device__ __forceinline__ void wait_asynccnt0() {
  asm volatile("s_wait_asynccnt 0" ::: "memory");
}

// ---------------------------------------------------------------------------
// Kernel 0a: xb = bf16(x)  (coalesced, 8 elems/thread)
// ---------------------------------------------------------------------------
__global__ __launch_bounds__(256) void k_cvt_x(const float* __restrict__ x,
                                               bf16* __restrict__ xb) {
  size_t i = ((size_t)blockIdx.x * 256 + threadIdx.x) * 8;
  v4f f0 = *(const v4f*)(x + i);
  v4f f1 = *(const v4f*)(x + i + 4);
  v8bf o;
#pragma unroll
  for (int e = 0; e < 4; ++e) { o[e] = (bf16)f0[e]; o[4 + e] = (bf16)f1[e]; }
  *(v8bf*)(xb + i) = o;
}

// ---------------------------------------------------------------------------
// Kernel 0b: Wt[fo][k] = (bf16) W[fo][k]  (so h-GEMM B-frags are contiguous)
// ---------------------------------------------------------------------------
__global__ __launch_bounds__(256) void k_wt(const float* __restrict__ W,
                                            bf16* __restrict__ Wt) {
  int idx = blockIdx.x * 256 + threadIdx.x;
  int k  = idx >> 9;
  int fo = idx & 511;
  Wt[(size_t)fo * FIN + k] = (bf16)W[(size_t)fo * FIN + k];
}

// 16x32 bf16 A-fragment from a row-major bf16 row pointer.
// ISA A-layout: lane holds row M=lane&15; K chunks (lane>>4)*8+{0..7} in
// elements 0..7 and +16 in elements 8..15.
__device__ __forceinline__ v16bf make_afrag(const bf16* __restrict__ base) {
  v8bf lo = *(const v8bf*)(base);
  v8bf hi = *(const v8bf*)(base + 16);
  v16bf r;
#pragma unroll
  for (int e = 0; e < 8; ++e) { r[e] = lo[e]; r[8 + e] = hi[e]; }
  return r;
}

// ---------------------------------------------------------------------------
// Kernel 1: h = x @ W^T via WMMA bf16; store hb (bf16), fuse p=h@a1, q=h@a2.
// Workgroup tile 128x64 (8 waves, each 32x32 = 2x2 WMMA accumulators).
// ---------------------------------------------------------------------------
__device__ __forceinline__ void epilogue_tile(v8f c, int gM, int gN, int lane,
                                              bf16* __restrict__ hb,
                                              float* __restrict__ p,
                                              float* __restrict__ q,
                                              const float* __restrict__ a1v,
                                              const float* __restrict__ a2v) {
  const int hi  = lane >> 4;     // C layout: row = vgpr + 8*hi
  const int col = lane & 15;     // C layout: lane(lo4) = column
  float a1c = a1v[gN + col];
  float a2c = a2v[gN + col];
  float pp[8], qq[8];
#pragma unroll
  for (int r = 0; r < 8; ++r) {
    int m = gM + hi * 8 + r;
    hb[(size_t)m * FOUT + gN + col] = (bf16)c[r];
    pp[r] = c[r] * a1c;
    qq[r] = c[r] * a2c;
  }
#pragma unroll
  for (int msk = 1; msk < 16; msk <<= 1) {
#pragma unroll
    for (int r = 0; r < 8; ++r) {
      pp[r] += __shfl_xor(pp[r], msk, 32);
      qq[r] += __shfl_xor(qq[r], msk, 32);
    }
  }
  if (col == 0) {
#pragma unroll
    for (int r = 0; r < 8; ++r) {
      atomicAdd(&p[gM + hi * 8 + r], pp[r]);
      atomicAdd(&q[gM + hi * 8 + r], qq[r]);
    }
  }
}

__global__ __launch_bounds__(256) void k_gemm_h(const bf16* __restrict__ xb,
                                                const bf16* __restrict__ Wt,
                                                const float* __restrict__ avec,
                                                bf16* __restrict__ hb,
                                                float* __restrict__ p,
                                                float* __restrict__ q) {
  const int lane = threadIdx.x & 31;
  const int wave = threadIdx.x >> 5;
  const int wm = wave >> 1;            // 0..3 (M)
  const int wn = wave & 1;             // 0..1 (N)
  const int mt = blockIdx.x & 63;      // 64 M-tiles of 128
  const int nt = blockIdx.x >> 6;      // 8 N-tiles of 64
  const int tileM = mt * 128 + wm * 32;
  const int tileN = nt * 64 + wn * 32;
  const int row = lane & 15;
  const int kb  = (lane >> 4) * 8;

  const v8f vz = {0.f, 0.f, 0.f, 0.f, 0.f, 0.f, 0.f, 0.f};
  v8f c00 = vz, c01 = vz, c10 = vz, c11 = vz;

  for (int k0 = 0; k0 < FIN; k0 += 32) {
    v16bf a0 = make_afrag(xb + (size_t)(tileM + row) * FIN + k0 + kb);
    v16bf a1 = make_afrag(xb + (size_t)(tileM + 16 + row) * FIN + k0 + kb);
    // B frag: lane l holds column n=tileN+(l&15), K in the same order as A
    const bf16* bptr = Wt + (size_t)(tileN + row) * FIN + k0 + kb;
    v16bf b0 = make_afrag(bptr);
    v16bf b1 = make_afrag(bptr + (size_t)16 * FIN);
    c00 = __builtin_amdgcn_wmma_f32_16x16x32_bf16(false, a0, false, b0, (short)0, c00, false, false);
    c01 = __builtin_amdgcn_wmma_f32_16x16x32_bf16(false, a0, false, b1, (short)0, c01, false, false);
    c10 = __builtin_amdgcn_wmma_f32_16x16x32_bf16(false, a1, false, b0, (short)0, c10, false, false);
    c11 = __builtin_amdgcn_wmma_f32_16x16x32_bf16(false, a1, false, b1, (short)0, c11, false, false);
  }
  const float* a1v = avec;
  const float* a2v = avec + FOUT;
  epilogue_tile(c00, tileM,      tileN,      lane, hb, p, q, a1v, a2v);
  epilogue_tile(c01, tileM,      tileN + 16, lane, hb, p, q, a1v, a2v);
  epilogue_tile(c10, tileM + 16, tileN,      lane, hb, p, q, a1v, a2v);
  epilogue_tile(c11, tileM + 16, tileN + 16, lane, hb, p, q, a1v, a2v);
}

// ---------------------------------------------------------------------------
// Kernel 2: qmax = max_j q[j]
// ---------------------------------------------------------------------------
__global__ __launch_bounds__(1024) void k_qmax(const float* __restrict__ q,
                                               float* __restrict__ qmax) {
  __shared__ float red[32];
  float m = -3.402823466e+38f;
  for (int i = threadIdx.x; i < NROWS; i += 1024) m = fmaxf(m, q[i]);
#pragma unroll
  for (int s = 16; s >= 1; s >>= 1) m = fmaxf(m, __shfl_xor(m, s, 32));
  if ((threadIdx.x & 31) == 0) red[threadIdx.x >> 5] = m;
  __syncthreads();
  if (threadIdx.x < 32) {
    float v = red[threadIdx.x];
#pragma unroll
    for (int s = 16; s >= 1; s >>= 1) v = fmaxf(v, __shfl_xor(v, s, 32));
    if (threadIdx.x == 0) *qmax = v;
  }
}

// ---------------------------------------------------------------------------
// Kernel 3: Z_i = sum_j exp(LR(p_i+q_j) - m_i), m_i = LR(p_i + qmax).
// ---------------------------------------------------------------------------
__global__ __launch_bounds__(256) void k_z(const float* __restrict__ p,
                                           const float* __restrict__ q,
                                           const float* __restrict__ qmaxp,
                                           float* __restrict__ Z) {
  const int row = (blockIdx.x & 31) * 256 + threadIdx.x;
  const int j0  = (blockIdx.x >> 5) * 512;
  const float pi = p[row];
  const float mi = lrelu(pi + *qmaxp);
  float sum = 0.f;
  for (int j = j0; j < j0 + 512; ++j) {
    float s = lrelu(pi + q[j]);
    sum += __expf(s - mi);
  }
  atomicAdd(&Z[row], sum);
}

// ---------------------------------------------------------------------------
// Kernel 4: partial out = attn[:, jrange] @ h[jrange, :], streaming j in
// steps of 32. Block = 256 threads (8 waves): 4 row-blocks x 2 col-splits;
// each wave 16 rows x 256 cols (16 v8f accumulators). The 32x512 bf16 hb
// j-tile is DMA'd to LDS once per block (async, double-buffered) and shared
// by all 8 waves. 1/Z folded into exp weights => pure-store epilogue.
// ---------------------------------------------------------------------------
__global__ __launch_bounds__(256) void k_av(const bf16* __restrict__ hb,
                                            const float* __restrict__ p,
                                            const float* __restrict__ q,
                                            const float* __restrict__ Z,
                                            const float* __restrict__ qmaxp,
                                            float* __restrict__ outp) {
  __shared__ bf16 tile[2][JTILE][LDS_STRIDE];

  const int tid  = threadIdx.x;
  const int lane = tid & 31;
  const int wave = tid >> 5;           // 0..7
  const int rowblk = wave >> 1;        // 0..3
  const int colspl = wave & 1;         // 0..1
  const int jhalf = blockIdx.x & 1;
  const int m0 = (blockIdx.x >> 1) * ROWS_BLK + rowblk * 16;
  const int f0 = colspl * 256;
  const int jbeg = jhalf * (NROWS / JSPLIT);
  const int row = lane & 15;
  const int kb  = (lane >> 4) * 8;

  // DMA indexing: 256 threads move 32x512 bf16 (32KB) => 128B/thread
  const int ldr = tid >> 3;            // 0..31: j-row within tile
  const int ldc = tid & 7;             // 0..7 : 128B chunk within row
  const unsigned lb0 = (unsigned)(size_t)&tile[0][ldr][ldc * 64];
  const unsigned lb1 = (unsigned)(size_t)&tile[1][ldr][ldc * 64];

  auto issue = [&](int j0, unsigned lbase) {
    const bf16* g = hb + (size_t)(j0 + ldr) * FOUT + ldc * 64;
#pragma unroll
    for (int k = 0; k < 8; ++k)
      async_ld_b128(lbase + 16u * k, g + 8 * k);
  };

  const float pi   = p[m0 + row];
  const float mi   = lrelu(pi + *qmaxp);
  const float invZ = 1.0f / Z[m0 + row];

  const v8f vz = {0.f, 0.f, 0.f, 0.f, 0.f, 0.f, 0.f, 0.f};
  v8f acc[16];
#pragma unroll
  for (int t = 0; t < 16; ++t) acc[t] = vz;

  issue(jbeg, lb0);
  const int niter = (NROWS / JSPLIT) / JTILE;   // 128
  for (int it = 0; it < niter; ++it) {
    const int j0 = jbeg + it * JTILE;
    const int buf = it & 1;
    wait_asynccnt0();          // this wave's DMA for `buf` has landed
    __syncthreads();           // => every wave's slice has landed
    if (it + 1 < niter) issue(j0 + JTILE, (buf ^ 1) ? lb1 : lb0);

    // exp-weight A fragment (16x32) per the ISA A layout
    const float* qp = q + j0 + kb;
    v4f q0 = *(const v4f*)(qp);
    v4f q1 = *(const v4f*)(qp + 4);
    v4f q2 = *(const v4f*)(qp + 16);
    v4f q3 = *(const v4f*)(qp + 20);
    v16bf afrag;
#pragma unroll
    for (int e = 0; e < 4; ++e) {
      afrag[e]      = (bf16)(__expf(lrelu(pi + q0[e]) - mi) * invZ);
      afrag[4 + e]  = (bf16)(__expf(lrelu(pi + q1[e]) - mi) * invZ);
      afrag[8 + e]  = (bf16)(__expf(lrelu(pi + q2[e]) - mi) * invZ);
      afrag[12 + e] = (bf16)(__expf(lrelu(pi + q3[e]) - mi) * invZ);
    }

    // B frags from LDS: lane l holds column f0+16t+(l&15), K in A's order
#pragma unroll
    for (int t = 0; t < 16; ++t) {
      v16bf b;
#pragma unroll
      for (int e = 0; e < 8; ++e) { b[e] = tile[buf][kb + e][f0 + 16 * t + row]; b[8 + e] = tile[buf][kb + 16 + e][f0 + 16 * t + row]; }
      acc[t] = __builtin_amdgcn_wmma_f32_16x16x32_bf16(false, afrag, false, b, (short)0, acc[t], false, false);
    }
    __syncthreads();           // all waves done reading `buf` before overwrite
  }

  float* o = outp + (size_t)jhalf * NROWS * FOUT;
  const int hi  = lane >> 4;
  const int col = lane & 15;
#pragma unroll
  for (int t = 0; t < 16; ++t) {
#pragma unroll
    for (int r = 0; r < 8; ++r) {
      o[(size_t)(m0 + hi * 8 + r) * FOUT + f0 + 16 * t + col] = acc[t][r];
    }
  }
}

// ---------------------------------------------------------------------------
// Kernel 5: out = out0 + out1 (combine j-split partials)
// ---------------------------------------------------------------------------
__global__ __launch_bounds__(256) void k_add(const float* __restrict__ o0,
                                             const float* __restrict__ o1,
                                             float* __restrict__ out) {
  size_t i = ((size_t)blockIdx.x * 256 + threadIdx.x) * 4;
  v4f a = *(const v4f*)(o0 + i);
  v4f b = *(const v4f*)(o1 + i);
  a = a + b;
  *(v4f*)(out + i) = a;
}

// ---------------------------------------------------------------------------
extern "C" void kernel_launch(void* const* d_in, const int* in_sizes, int n_in,
                              void* d_out, int out_size, void* d_ws, size_t ws_size,
                              hipStream_t stream) {
  (void)in_sizes; (void)n_in; (void)out_size; (void)ws_size;
  const float* x = (const float*)d_in[0];   // 8192 x 512
  const float* W = (const float*)d_in[1];   // 512 x 512
  const float* a = (const float*)d_in[2];   // 1 x 1024
  float* out = (float*)d_out;               // 8192 x 512 (f32)

  char* ws = (char*)d_ws;
  size_t off = 0;
  bf16*  Wt   = (bf16*)(ws + off);  off += (size_t)FIN * FOUT * sizeof(bf16);    // 0.5 MB
  bf16*  xb   = (bf16*)(ws + off);  off += (size_t)NROWS * FIN * sizeof(bf16);   // 8 MB
  bf16*  hb   = (bf16*)(ws + off);  off += (size_t)NROWS * FOUT * sizeof(bf16);  // 8 MB
  float* p    = (float*)(ws + off); off += (size_t)NROWS * sizeof(float);
  float* q    = (float*)(ws + off); off += (size_t)NROWS * sizeof(float);
  float* Z    = (float*)(ws + off); off += (size_t)NROWS * sizeof(float);
  float* qmax = (float*)(ws + off); off += 256;
  float* oprt = (float*)(ws + off); off += (size_t)JSPLIT * NROWS * FOUT * sizeof(float); // 32 MB

  hipMemsetAsync(p, 0, (size_t)3 * NROWS * sizeof(float), stream);  // p,q,Z contiguous

  k_cvt_x <<<(NROWS * FIN) / (256 * 8), 256, 0, stream>>>(x, xb);
  k_wt    <<<(FIN * FOUT) / 256,        256, 0, stream>>>(W, Wt);
  k_gemm_h<<<dim3(64 * 8),              256, 0, stream>>>(xb, Wt, a, hb, p, q);
  k_qmax  <<<1,                        1024, 0, stream>>>(q, qmax);
  k_z     <<<dim3(32 * 16),             256, 0, stream>>>(p, q, qmax, Z);
  k_av    <<<dim3((NROWS / ROWS_BLK) * JSPLIT), 256, 0, stream>>>(hb, p, q, Z, qmax, oprt);
  k_add   <<<(NROWS * FOUT) / (256 * 4), 256, 0, stream>>>(oprt, oprt + (size_t)NROWS * FOUT, out);
}
